// MultiQueryAttention_90314572300555
// MI455X (gfx1250) — matrix operand order repaired and checked by reading, not compiled, not run on hardware
//
#include <hip/hip_runtime.h>
#include <hip/hip_bf16.h>
#include <stdint.h>

// ---------------------------------------------------------------------------
// MQA forward for MI455X (gfx1250, wave32, WMMA).
// fp32 -> bf16 cast, WMMA bf16 GEMMs (fp32 accum), flash attention.
// Round 3: double-buffered LDS tiles, one barrier per iteration, async
// global->LDS DMA (ASYNCcnt) overlapped with the WMMA compute of the
// previous tile.
// ---------------------------------------------------------------------------

#define D_MODEL  2048
#define N_HEADS  16
#define HEAD_DIM 128
#define BATCH    2
#define SEQ      2048
#define M_TOTAL  (BATCH * SEQ)   // 4096 token rows

typedef __bf16 bf16_t;
typedef __attribute__((ext_vector_type(16))) __bf16 v16bf;
typedef __attribute__((ext_vector_type(8)))  float  v8f;

__device__ __forceinline__ v8f wmma_bf16f32(v16bf a, v16bf b, v8f c) {
  // D = A(16x32 bf16) * B(32x16 bf16) + C(16x16 f32)
  return __builtin_amdgcn_wmma_f32_16x16x32_bf16(false, a, false, b,
                                                 (short)0, c, false, false);
}

// Async global->LDS 16B copy (CDNA5 GLOBAL_LOAD_ASYNC_TO_LDS_B128).
// LDS destination offset comes from the low 32 bits of the generic pointer
// (addrspace(3) aperture: [31:0] = LDS offset). Tracked by ASYNCcnt.
__device__ __forceinline__ void async_load_b128(void* lds_dst,
                                                const void* gsrc) {
  const uint32_t loff = (uint32_t)(uintptr_t)lds_dst;
  asm volatile("global_load_async_to_lds_b128 %0, %1, off"
               :: "v"(loff), "v"(gsrc)
               : "memory");
}
__device__ __forceinline__ void wait_asynccnt0() {
  asm volatile("s_wait_asynccnt 0x0" ::: "memory");
}

// ---------------------------------------------------------------------------
// Fragment gathers from LDS following ISA 7.12.2 (wave32):
//  A 16x32 bf16: lane = m (0..15) [+16 selects other K-half];
//    VGPR v, pair p -> k = (v<4 ? 0 : 16) + (v%4)*2 + half*8 + p
//  B 32x16 bf16: lane = n (0..15) [+16 selects K 8..15, 24..31];
//    VGPR v, pair p -> k = (v<4 ? 0 : 16) + (v%4)*2 + half*8 + p
//  C/D 16x16 f32: lane = n (0..15), element e -> row = e + 8*(lane>=16)
// ---------------------------------------------------------------------------
__device__ __forceinline__ v16bf load_a_frag(const bf16_t* base, int row0,
                                             int stride, int lane) {
  const int m = lane & 15, half = lane >> 4;
  const bf16_t* rp = base + (size_t)(row0 + m) * stride;
  v16bf r;
#pragma unroll
  for (int v = 0; v < 8; ++v) {
    const int k = ((v < 4) ? 0 : 16) + (v & 3) * 2 + half * 8;
    r[2 * v]     = rp[k];
    r[2 * v + 1] = rp[k + 1];
  }
  return r;
}

// B fragment from an [n][k] (transposed) LDS tile: contiguous 32-bit reads.
__device__ __forceinline__ v16bf load_b_fragT(const bf16_t* base, int col0,
                                              int stride, int lane) {
  const int n = lane & 15, half = lane >> 4;
  const bf16_t* cp = base + (size_t)(col0 + n) * stride;
  v16bf r;
#pragma unroll
  for (int v = 0; v < 8; ++v) {
    const int k = ((v < 4) ? 0 : 16) + (v & 3) * 2 + half * 8;
    r[2 * v]     = cp[k];
    r[2 * v + 1] = cp[k + 1];
  }
  return r;
}

// ---------------------------------------------------------------------------
// fp32 -> bf16 cast (4 elements/thread; all sizes are multiples of 4)
// ---------------------------------------------------------------------------
__global__ void cvt_f32_bf16(const float* __restrict__ in,
                             bf16_t* __restrict__ out, int n) {
  int i = (blockIdx.x * blockDim.x + threadIdx.x) * 4;
  if (i + 3 < n) {
    const float4 v = *(const float4*)(in + i);
    out[i]     = (bf16_t)v.x;
    out[i + 1] = (bf16_t)v.y;
    out[i + 2] = (bf16_t)v.z;
    out[i + 3] = (bf16_t)v.w;
  }
}

// ---------------------------------------------------------------------------
// Tiled GEMM: C[M,N] = A[M,K](bf16,rm) * B[K,N](bf16,rm) + bias[N]
// Block tile 128x128, BK=32, 256 threads = 8 waves (2x4), wave tile 64x32,
// each wave holds 4x2 accumulators of 16x16 f32 (WMMA 16x16x32 bf16).
// Double-buffered LDS; A via async DMA, B staged transposed; one barrier
// per K-step (rendezvous orders buf[nxt] writes after prior reads).
// ---------------------------------------------------------------------------
template <bool OUT_BF16>
__global__ __launch_bounds__(256)
void gemm_bf16(const bf16_t* __restrict__ A, const bf16_t* __restrict__ B,
               const float* __restrict__ bias, void* __restrict__ Cout,
               int M, int N, int K) {
  constexpr int BM = 128, BN = 128, BK = 32;
  __shared__ __align__(16) bf16_t sA[2][BM * BK];   // [m][k] row-major
  __shared__ __align__(16) bf16_t sBt[2][BN * BK];  // [n][k] transposed

  const int tid  = threadIdx.x;
  const int lane = tid & 31;
  const int wave = tid >> 5;
  const int wm   = wave >> 2;  // 0..1
  const int wn   = wave & 3;   // 0..3
  const int m0   = blockIdx.y * BM;
  const int n0   = blockIdx.x * BN;

  auto stage = [&](int k0, int buf) {
#pragma unroll
    for (int it = 0; it < 2; ++it) {
      const int off = (tid + it * 256) * 8;
      {  // A tile: async global->LDS DMA, row-major
        const int r = off / BK, c = off % BK;
        async_load_b128(&sA[buf][r * BK + c],
                        A + (size_t)(m0 + r) * K + k0 + c);
      }
      {  // B tile: load 16B, scatter transposed into sBt[n][k]
        const int r = off / BN, c = off % BN;
        if (k0 + BK < K)  // gfx1250 global_prefetch_b8 for tile after next
          __builtin_prefetch(B + (size_t)(k0 + BK + r) * N + n0 + c, 0, 1);
        const uint4 v = *(const uint4*)(B + (size_t)(k0 + r) * N + n0 + c);
        const bf16_t* e = (const bf16_t*)&v;
#pragma unroll
        for (int j = 0; j < 8; ++j) sBt[buf][(c + j) * BK + r] = e[j];
      }
    }
  };

  const v8f zero = {};
  v8f acc[4][2];
#pragma unroll
  for (int i = 0; i < 4; ++i)
#pragma unroll
    for (int j = 0; j < 2; ++j) acc[i][j] = zero;

  stage(0, 0);
  for (int k0 = 0, t = 0; k0 < K; k0 += BK, ++t) {
    const int cur = t & 1;
    wait_asynccnt0();      // buf[cur] DMA complete (this wave)
    __syncthreads();       // ... and for all waves; orders buf reuse
    if (k0 + BK < K) stage(k0 + BK, cur ^ 1);  // DMA overlaps compute below

    v16bf af[4];
#pragma unroll
    for (int i = 0; i < 4; ++i)
      af[i] = load_a_frag(sA[cur], wm * 64 + i * 16, BK, lane);
    v16bf bfg[2];
#pragma unroll
    for (int j = 0; j < 2; ++j)
      bfg[j] = load_b_fragT(sBt[cur], wn * 32 + j * 16, BK, lane);
#pragma unroll
    for (int i = 0; i < 4; ++i)
#pragma unroll
      for (int j = 0; j < 2; ++j)
        acc[i][j] = wmma_bf16f32(af[i], bfg[j], acc[i][j]);
  }

  // Epilogue: C/D layout -> global (M,N exact multiples of tile; no guards)
  const int ln = lane & 15, half = lane >> 4;
#pragma unroll
  for (int i = 0; i < 4; ++i)
#pragma unroll
    for (int j = 0; j < 2; ++j) {
      const int nn = n0 + wn * 32 + j * 16 + ln;
      const float bv = bias ? bias[nn] : 0.f;
#pragma unroll
      for (int e = 0; e < 8; ++e) {
        const int mm = m0 + wm * 64 + i * 16 + e + half * 8;
        const float val = acc[i][j][e] + bv;
        if (OUT_BF16)
          ((bf16_t*)Cout)[(size_t)mm * N + nn] = (bf16_t)val;
        else
          ((float*)Cout)[(size_t)mm * N + nn] = val;
      }
    }
}

// ---------------------------------------------------------------------------
// Flash attention (multi-query: one shared K/V head).
// Block = (b, h, 128 q rows) = 8 waves x 16 q rows. All 8 waves share the
// double-buffered 32x128 K tile (async DMA, row-major) and V tile (staged
// transposed so P*V B-frags are contiguous b128 reads). One barrier per key
// tile; online softmax with 16-lane butterfly reductions; P round-trips a
// per-wave LDS tile (C-layout -> A-layout, ordered by DScnt).
// ---------------------------------------------------------------------------
__global__ __launch_bounds__(256)
void attn_fa(const bf16_t* __restrict__ Q, const bf16_t* __restrict__ Kc,
             const bf16_t* __restrict__ Vc, const uint8_t* __restrict__ mask,
             bf16_t* __restrict__ ctx) {
  constexpr int KT = 32;  // keys per iteration
  __shared__ __align__(16) bf16_t sK[2][KT * HEAD_DIM];   // [key][d]
  __shared__ __align__(16) bf16_t sVt[2][HEAD_DIM * KT];  // [n][k] transposed
  __shared__ __align__(16) bf16_t sP[8][16 * KT];

  const int tid = threadIdx.x, lane = tid & 31, wave = tid >> 5;
  const int b = blockIdx.z, h = blockIdx.y;
  const int qrow0 = blockIdx.x * 128 + wave * 16;
  const int ln = lane & 15, half = lane >> 4;

  auto stageKV = [&](int kt, int buf) {
#pragma unroll
    for (int it = 0; it < 2; ++it) {
      const int off = (tid + it * 256) * 8;
      const int r = off / HEAD_DIM, c = off % HEAD_DIM;
      async_load_b128(&sK[buf][r * HEAD_DIM + c],
                      Kc + (size_t)(b * SEQ + kt + r) * HEAD_DIM + c);
      const uint4 v =
          *(const uint4*)(Vc + (size_t)(b * SEQ + kt + r) * HEAD_DIM + c);
      const bf16_t* e = (const bf16_t*)&v;
#pragma unroll
      for (int j = 0; j < 8; ++j) sVt[buf][(c + j) * KT + r] = e[j];
    }
  };

  // Q fragments: 16x128 as 4 K-chunks of 32 (A layout, straight from global)
  v16bf qf[4];
  {
    const bf16_t* qbase =
        Q + (size_t)(b * SEQ + qrow0) * D_MODEL + h * HEAD_DIM;
#pragma unroll
    for (int c = 0; c < 4; ++c)
      qf[c] = load_a_frag(qbase + c * 32, 0, D_MODEL, lane);
  }

  const v8f zero = {};
  v8f oacc[8];
#pragma unroll
  for (int vb = 0; vb < 8; ++vb) oacc[vb] = zero;
  float rm[8], rl[8];
#pragma unroll
  for (int e = 0; e < 8; ++e) { rm[e] = -1e30f; rl[e] = 0.f; }

  const float scale = 0.08838834764831845f;  // 1/sqrt(128)

  stageKV(0, 0);
  for (int kt = 0, t = 0; kt < SEQ; kt += KT, ++t) {
    const int cur = t & 1;
    wait_asynccnt0();
    __syncthreads();
    if (kt + KT < SEQ) stageKV(kt + KT, cur ^ 1);  // DMA overlaps compute

    // scores S = Q * K^T : two 16x16 f32 tiles (keys 0..15, 16..31)
    v8f s0 = zero, s1 = zero;
#pragma unroll
    for (int c = 0; c < 4; ++c) {
      // B[dd][n] = K[key = n][d = c*32+dd]; sK row-major -> contiguous reads
      v16bf bk0 = load_b_fragT(sK[cur] + c * 32, 0, HEAD_DIM, lane);
      v16bf bk1 = load_b_fragT(sK[cur] + c * 32, 16, HEAD_DIM, lane);
      s0 = wmma_bf16f32(qf[c], bk0, s0);
      s1 = wmma_bf16f32(qf[c], bk1, s1);
    }

    // One key column per lane -> one mask load per tile per lane
    const float madd0 = mask[b * SEQ + kt + ln] ? 0.f : -1e9f;
    const float madd1 = mask[b * SEQ + kt + 16 + ln] ? 0.f : -1e9f;

    float corr[8];
#pragma unroll
    for (int e = 0; e < 8; ++e) {
      float a = s0[e] * scale + madd0;
      float bb = s1[e] * scale + madd1;
      float mx = fmaxf(a, bb);
#pragma unroll
      for (int d = 8; d >= 1; d >>= 1) mx = fmaxf(mx, __shfl_xor(mx, d, 16));
      const float nm = fmaxf(rm[e], mx);
      corr[e] = __expf(rm[e] - nm);
      rm[e] = nm;
      const float p0 = __expf(a - nm);
      const float p1 = __expf(bb - nm);
      float rs = p0 + p1;
#pragma unroll
      for (int d = 8; d >= 1; d >>= 1) rs += __shfl_xor(rs, d, 16);
      rl[e] = rl[e] * corr[e] + rs;
      // stash P (C layout row = e + 8*half) into this wave's LDS tile
      sP[wave][(e + 8 * half) * KT + ln]      = (bf16_t)p0;
      sP[wave][(e + 8 * half) * KT + 16 + ln] = (bf16_t)p1;
    }
#pragma unroll
    for (int vb = 0; vb < 8; ++vb)
#pragma unroll
      for (int e = 0; e < 8; ++e) oacc[vb][e] *= corr[e];

    // O += P(16x32) * V(32x128) ; same-wave LDS RAW ordered via DScnt.
    // B[kk][n] = V[kt+kk][vb*16+n] = sVt[(vb*16+n)*KT + kk] (contiguous).
    v16bf pa = load_a_frag(sP[wave], 0, KT, lane);
#pragma unroll
    for (int vb = 0; vb < 8; ++vb) {
      v16bf bv = load_b_fragT(sVt[cur], vb * 16, KT, lane);
      oacc[vb] = wmma_bf16f32(pa, bv, oacc[vb]);
    }
  }

  // Normalize and write context (bf16) for the output projection
#pragma unroll
  for (int e = 0; e < 8; ++e) rl[e] = 1.f / rl[e];
#pragma unroll
  for (int vb = 0; vb < 8; ++vb)
#pragma unroll
    for (int e = 0; e < 8; ++e) {
      const int srow = qrow0 + e + 8 * half;
      ctx[(size_t)(b * SEQ + srow) * D_MODEL + h * HEAD_DIM + vb * 16 + ln] =
          (bf16_t)(oacc[vb][e] * rl[e]);
    }
}

// ---------------------------------------------------------------------------
// Host-side orchestration.
// Inputs: 0:x 1:Wq 2:bq 3:Wk 4:bk 5:Wv 6:bv 7:Wo 8:bo 9:cache_k 10:cache_v
//         11:mask 12:start_pos (start_pos==0; cache is not an output)
// Workspace layout (bf16): x, Wq, Wk, Wv, Wo, q, k, v, ctx  (~68 MB)
// ---------------------------------------------------------------------------
extern "C" void kernel_launch(void* const* d_in, const int* in_sizes, int n_in,
                              void* d_out, int out_size, void* d_ws,
                              size_t ws_size, hipStream_t stream) {
  (void)in_sizes; (void)n_in; (void)out_size; (void)ws_size;
  const float* x    = (const float*)d_in[0];
  const float* Wq   = (const float*)d_in[1];
  const float* bq   = (const float*)d_in[2];
  const float* Wk   = (const float*)d_in[3];
  const float* bk   = (const float*)d_in[4];
  const float* Wv   = (const float*)d_in[5];
  const float* bv   = (const float*)d_in[6];
  const float* Wo   = (const float*)d_in[7];
  const float* bo   = (const float*)d_in[8];
  const uint8_t* mask = (const uint8_t*)d_in[11];
  float* out = (float*)d_out;

  bf16_t* p = (bf16_t*)d_ws;
  bf16_t* xb   = p; p += (size_t)M_TOTAL * D_MODEL;
  bf16_t* wqb  = p; p += (size_t)D_MODEL * D_MODEL;
  bf16_t* wkb  = p; p += (size_t)D_MODEL * HEAD_DIM;
  bf16_t* wvb  = p; p += (size_t)D_MODEL * HEAD_DIM;
  bf16_t* wob  = p; p += (size_t)D_MODEL * D_MODEL;
  bf16_t* qb   = p; p += (size_t)M_TOTAL * D_MODEL;
  bf16_t* kb   = p; p += (size_t)M_TOTAL * HEAD_DIM;
  bf16_t* vbuf = p; p += (size_t)M_TOTAL * HEAD_DIM;
  bf16_t* ctxb = p;

  auto cvt = [&](const float* src, bf16_t* dst, size_t n) {
    cvt_f32_bf16<<<(unsigned)(n / 1024), 256, 0, stream>>>(src, dst, (int)n);
  };
  cvt(x,  xb,  (size_t)M_TOTAL * D_MODEL);
  cvt(Wq, wqb, (size_t)D_MODEL * D_MODEL);
  cvt(Wk, wkb, (size_t)D_MODEL * HEAD_DIM);
  cvt(Wv, wvb, (size_t)D_MODEL * HEAD_DIM);
  cvt(Wo, wob, (size_t)D_MODEL * D_MODEL);

  const dim3 blk(256);
  // Q/K/V projections
  gemm_bf16<true><<<dim3(D_MODEL / 128, M_TOTAL / 128), blk, 0, stream>>>(
      xb, wqb, bq, qb, M_TOTAL, D_MODEL, D_MODEL);
  gemm_bf16<true><<<dim3(HEAD_DIM / 128, M_TOTAL / 128), blk, 0, stream>>>(
      xb, wkb, bk, kb, M_TOTAL, HEAD_DIM, D_MODEL);
  gemm_bf16<true><<<dim3(HEAD_DIM / 128, M_TOTAL / 128), blk, 0, stream>>>(
      xb, wvb, bv, vbuf, M_TOTAL, HEAD_DIM, D_MODEL);

  // Attention
  attn_fa<<<dim3(SEQ / 128, N_HEADS, BATCH), blk, 0, stream>>>(
      qb, kb, vbuf, mask, ctxb);

  // Output projection (fp32 out + bias)
  gemm_bf16<false><<<dim3(D_MODEL / 128, M_TOTAL / 128), blk, 0, stream>>>(
      ctxb, wob, bo, out, M_TOTAL, D_MODEL, D_MODEL);
}
